// MultiHeadAttention_87428354278038
// MI455X (gfx1250) — matrix operand order repaired and checked by reading, not compiled, not run on hardware
//
#include <hip/hip_runtime.h>
#include <hip/hip_bf16.h>

// ---------------------------------------------------------------------------
// MHA on gfx1250 (CDNA5): bf16 WMMA GEMMs + flash-attention.
//   D_MODEL=1024, H=16, Dh=64, B=4, S=2048, M = B*S = 8192.
// LDS tiles use fragment-contiguous layouts (ds_load_2addr_b64 pairs) with
// padded row strides; softmax reductions are DPP16 butterflies; the flash
// K-tile staging uses CDNA5 async LDS DMA (global_load_async_to_lds_b128,
// tracked by ASYNCcnt).
// ---------------------------------------------------------------------------

typedef __attribute__((ext_vector_type(16))) __bf16 v16bf;
typedef __attribute__((ext_vector_type(8)))  float  v8f;

static __device__ __forceinline__ v8f wmma_bf16(v16bf a, v16bf b, v8f c) {
    return __builtin_amdgcn_wmma_f32_16x16x32_bf16(
        /*neg_a=*/false, a, /*neg_b=*/false, b,
        /*c_mod=*/(short)0, c, /*reuse_a=*/false, /*reuse_b=*/false);
}

// --- DPP16 butterfly reductions over 16-lane groups (no LDS traffic) -------
template <int CTRL>
static __device__ __forceinline__ float dpp_xchg(float x) {
    return __builtin_bit_cast(
        float, __builtin_amdgcn_update_dpp(0, __builtin_bit_cast(int, x), CTRL,
                                           0xf, 0xf, true));
}
static __device__ __forceinline__ float row_max16(float x) {
    x = fmaxf(x, dpp_xchg<0x0B1>(x));  // quad_perm [1,0,3,2]  (xor 1)
    x = fmaxf(x, dpp_xchg<0x04E>(x));  // quad_perm [2,3,0,1]  (xor 2)
    x = fmaxf(x, dpp_xchg<0x141>(x));  // row_half_mirror      (8-group)
    x = fmaxf(x, dpp_xchg<0x140>(x));  // row_mirror           (16-group)
    return x;
}
static __device__ __forceinline__ float row_sum16(float x) {
    x += dpp_xchg<0x0B1>(x);
    x += dpp_xchg<0x04E>(x);
    x += dpp_xchg<0x141>(x);
    x += dpp_xchg<0x140>(x);
    return x;
}

// A-fragment (16x32 bf16, row-major source, leading dim lda), contiguous k order (matches B loaders):
// lanes 0-15: row M=lane,   K = k0+{0..15}
// lanes16-31: row M=lane-16,K = k0+{16..31}
static __device__ __forceinline__ v16bf load_a16(const __bf16* a, int lda,
                                                 int m0, int k0) {
    int lane = threadIdx.x & 31;
    const __bf16* p =
        a + (size_t)(m0 + (lane & 15)) * lda + k0 + ((lane >> 4) << 4);
    v16bf r;
#pragma unroll
    for (int j = 0; j < 8; ++j) r[j] = p[j];
#pragma unroll
    for (int j = 0; j < 8; ++j) r[j + 8] = p[8 + j];
    return r;
}

// B-fragment from a fragment-contiguous tile bs[n][k] (ldb-padded rows):
// lane holds col n = n0+(lane&15), K = {0..15} (lanes 0-15) / {16..31}.
static __device__ __forceinline__ v16bf load_bfrag(const __bf16* bs, int ldb,
                                                   int n0) {
    int lane = threadIdx.x & 31;
    const __bf16* p =
        bs + (size_t)(n0 + (lane & 15)) * ldb + ((lane >> 4) << 4);
    v16bf r;
#pragma unroll
    for (int j = 0; j < 16; ++j) r[j] = p[j];
    return r;
}

// B-fragment where the source tile is row-major [n][d] and the contraction
// runs along d (B = tile^T). Rows padded to ldk. Used for Q @ K^T.
static __device__ __forceinline__ v16bf load_bT16(const __bf16* kt, int ldk,
                                                  int n0, int d0) {
    int lane = threadIdx.x & 31;
    const __bf16* p =
        kt + (size_t)(n0 + (lane & 15)) * ldk + d0 + ((lane >> 4) << 4);
    v16bf r;
#pragma unroll
    for (int j = 0; j < 16; ++j) r[j] = p[j];
    return r;
}

// ---------------------------------------------------------------------------
// GEMM:  Y = X @ W^T + bias.   X:[M,K] (AT), W:[N,K] fp32 row-major.
// SPLIT=true  -> Y written bf16 as [B, H, S, Dh]   (QKV projections)
// SPLIT=false -> Y written OT  as [M, N] row-major (output projection)
// Tile: BM=BN=128, BK=32; 256 threads = 8 waves (2x4), 64x32 per wave.
// Global->register prefetch is software-pipelined against the WMMA stage.
// ---------------------------------------------------------------------------
template <typename AT, typename OT, bool SPLIT>
__global__ __launch_bounds__(256) void gemm_bias_kernel(
    const AT* __restrict__ X, const float* __restrict__ W,
    const float* __restrict__ bias, OT* __restrict__ Y,
    int M, int N, int K) {
    constexpr int BM = 128, BN = 128, BK = 32;
    constexpr int LDA = 36, LDB = 36;  // padded rows: conflict-free b128 reads
    __shared__ __bf16 As[BM * LDA];    // As[m][k]
    __shared__ __bf16 Bs[BN * LDB];    // Bs[n][k]  (fragment-contiguous)

    const int t    = threadIdx.x;
    const int lane = t & 31;
    const int wave = t >> 5;
    const int wm   = (wave >> 2) * 64;
    const int wn   = (wave & 3) * 32;
    const int row0 = blockIdx.y * BM;
    const int col0 = blockIdx.x * BN;

    // Each thread stages 16 contiguous K-elements of one A row / one W row.
    const int am = t >> 1, ak = (t & 1) << 4;
    const AT*    xrow = X + (size_t)(row0 + am) * K + ak;
    const float* wrow = W + (size_t)(col0 + am) * K + ak;

    v8f acc[4][2] = {};
    float ax[16], bx[16];
#pragma unroll
    for (int j = 0; j < 16; ++j) {
        ax[j] = (float)xrow[j];
        bx[j] = wrow[j];
    }

    for (int k0 = 0; k0 < K; k0 += BK) {
#pragma unroll
        for (int j = 0; j < 16; ++j) {
            As[am * LDA + ak + j] = (__bf16)ax[j];
            Bs[am * LDB + ak + j] = (__bf16)bx[j];
        }
        __syncthreads();

        if (k0 + BK < K) {  // prefetch next tile while WMMAs run
#pragma unroll
            for (int j = 0; j < 16; ++j) {
                ax[j] = (float)xrow[k0 + BK + j];
                bx[j] = wrow[k0 + BK + j];
            }
        }

        v16bf af[4], bfg[2];
#pragma unroll
        for (int i = 0; i < 4; ++i) af[i] = load_a16(As, LDA, wm + i * 16, 0);
#pragma unroll
        for (int j = 0; j < 2; ++j) bfg[j] = load_bfrag(Bs, LDB, wn + j * 16);
#pragma unroll
        for (int i = 0; i < 4; ++i)
#pragma unroll
            for (int j = 0; j < 2; ++j)
                acc[i][j] = wmma_bf16(af[i], bfg[j], acc[i][j]);
        __syncthreads();
    }

    // Epilogue. C layout: lane holds col n=lane&15, rows r + 8*(lane>>4).
#pragma unroll
    for (int i = 0; i < 4; ++i) {
#pragma unroll
        for (int j = 0; j < 2; ++j) {
            int   ng = col0 + wn + j * 16 + (lane & 15);
            float bv = bias[ng];
#pragma unroll
            for (int r = 0; r < 8; ++r) {
                int   mg = row0 + wm + i * 16 + r + ((lane >> 4) << 3);
                float v  = acc[i][j][r] + bv;
                if constexpr (SPLIT) {
                    int bb = mg >> 11, ss = mg & 2047;
                    int hh = ng >> 6,  dd = ng & 63;
                    Y[((((size_t)bb * 16 + hh) * 2048 + ss) << 6) + dd] = (OT)v;
                } else {
                    Y[(size_t)mg * N + ng] = (OT)v;
                }
            }
        }
    }
}

// ---------------------------------------------------------------------------
// Flash attention: one workgroup per (128-row q-block, head, batch).
// 8 waves, each owns a 16-row Q tile. K staged row-major via async LDS DMA
// (16 B per lane, ASYNCcnt); V staged transposed (fragment-contiguous for
// P@V) on the VALU path while the DMA is in flight.
// ---------------------------------------------------------------------------
__global__ __launch_bounds__(256) void flash_attn_kernel(
    const __bf16* __restrict__ Q, const __bf16* __restrict__ Kin,
    const __bf16* __restrict__ V, __bf16* __restrict__ Ctx) {
    constexpr int S = 2048, Dh = 64, H = 16, DM = 1024;
    constexpr int LDK = 68, LDV = 36, LDP = 36;  // padded rows
    __shared__ __bf16 Ks[32 * LDK];      // Ks[n][d]
    __shared__ __bf16 Vt[Dh * LDV];      // Vt[d][n]  (fragment-contiguous)
    __shared__ __bf16 Ps[8 * 16 * LDP];  // per-wave P tile (16x32)

    const int t    = threadIdx.x;
    const int lane = t & 31;
    const int wave = t >> 5;
    const int b    = blockIdx.z;
    const int h    = blockIdx.y;
    const int q0   = blockIdx.x * 128 + wave * 16;

    const size_t head_off = (((size_t)b * H + h) * S) * Dh;
    const __bf16* Qh = Q + head_off;
    const __bf16* Kh = Kin + head_off;
    const __bf16* Vh = V + head_off;

    const v16bf aq0 = load_a16(Qh, Dh, q0, 0);
    const v16bf aq1 = load_a16(Qh, Dh, q0, 32);

    v8f   o[4] = {};
    float mrun[8], lrun[8];
#pragma unroll
    for (int r = 0; r < 8; ++r) { mrun[r] = -1e30f; lrun[r] = 0.f; }

    __bf16*  Pw   = Ps + wave * 16 * LDP;
    const int prow = (lane >> 4) << 3;
    const int pcol = lane & 15;
    const int sn   = (t * 8) >> 6;   // staged row (8 elems/thread, 64/row)
    const int sd   = (t * 8) & 63;   // staged col (16B-aligned chunk)

    // LDS destination offset for this lane's async 16B K chunk. Generic
    // pointers to __shared__ carry the LDS offset in their low 32 bits.
    const unsigned ks_dst = (unsigned)(uintptr_t)&Ks[sn * LDK + sd];

    for (int n0 = 0; n0 < S; n0 += 32) {
        // Kick off the K-tile DMA: one global_load_async_to_lds_b128 per
        // lane (ASYNCcnt), then do the V transpose staging on the VALU path
        // while the DMA is in flight.
        {
            unsigned long long ksrc =
                (unsigned long long)(uintptr_t)(Kh + (size_t)n0 * Dh + t * 8);
            asm volatile("global_load_async_to_lds_b128 %0, %1, off"
                         :: "v"(ks_dst), "v"(ksrc)
                         : "memory");
        }
#pragma unroll
        for (int j = 0; j < 8; ++j)
            Vt[(sd + j) * LDV + sn] = Vh[(size_t)n0 * Dh + t * 8 + j];

        asm volatile("s_wait_asynccnt 0x0" ::: "memory");
        __syncthreads();

        // Scores: 16x32 block = two 16x16 f32 accumulators, K-dim = 64.
        v8f s0 = {}, s1 = {};
        s0 = wmma_bf16(aq0, load_bT16(Ks, LDK, 0, 0),   s0);
        s0 = wmma_bf16(aq1, load_bT16(Ks, LDK, 0, 32),  s0);
        s1 = wmma_bf16(aq0, load_bT16(Ks, LDK, 16, 0),  s1);
        s1 = wmma_bf16(aq1, load_bT16(Ks, LDK, 16, 32), s1);

        // Online softmax (scale = 1/sqrt(64)); DPP16 row reductions.
        float p0[8], p1[8], fac[8];
#pragma unroll
        for (int r = 0; r < 8; ++r) {
            float x0 = s0[r] * 0.125f;
            float x1 = s1[r] * 0.125f;
            float mx   = row_max16(fmaxf(x0, x1));
            float mnew = fmaxf(mrun[r], mx);
            fac[r] = __expf(mrun[r] - mnew);
            p0[r]  = __expf(x0 - mnew);
            p1[r]  = __expf(x1 - mnew);
            lrun[r] = lrun[r] * fac[r] + row_sum16(p0[r] + p1[r]);
            mrun[r] = mnew;
        }
#pragma unroll
        for (int dt = 0; dt < 4; ++dt)
#pragma unroll
            for (int r = 0; r < 8; ++r) o[dt][r] *= fac[r];

        // Re-shape P (C layout -> A layout) through per-wave LDS tile.
#pragma unroll
        for (int r = 0; r < 8; ++r) {
            Pw[(r + prow) * LDP + pcol]      = (__bf16)p0[r];
            Pw[(r + prow) * LDP + pcol + 16] = (__bf16)p1[r];
        }
        asm volatile("s_wait_dscnt 0x0" ::: "memory");
        v16bf ap = load_a16(Pw, LDP, 0, 0);

        // O += P (16x32) @ V (32x64): one WMMA per 16-wide d tile.
#pragma unroll
        for (int dt = 0; dt < 4; ++dt)
            o[dt] = wmma_bf16(ap, load_bfrag(Vt, LDV, dt * 16), o[dt]);

        __syncthreads();
    }

    // Normalize and write context in [B, S, D_MODEL] as bf16.
#pragma unroll
    for (int r = 0; r < 8; ++r) {
        float inv  = 1.0f / lrun[r];
        int   srow = q0 + r + prow;
#pragma unroll
        for (int dt = 0; dt < 4; ++dt) {
            int col = h * Dh + dt * 16 + pcol;
            Ctx[((size_t)b * S + srow) * DM + col] = (__bf16)(o[dt][r] * inv);
        }
    }
}

// ---------------------------------------------------------------------------
// Host-side launch. Inputs (fp32): query,key,value, Wq,bq, Wk,bk, Wv,bv, Wo,bo.
// Workspace: bf16 Q,K,V in [B,H,S,Dh] + bf16 context in [B,S,D] = 64 MiB.
// ---------------------------------------------------------------------------
extern "C" void kernel_launch(void* const* d_in, const int* in_sizes, int n_in,
                              void* d_out, int out_size, void* d_ws,
                              size_t ws_size, hipStream_t stream) {
    (void)in_sizes; (void)n_in; (void)out_size; (void)ws_size;

    constexpr int B = 4, S = 2048, DM = 1024, M = B * S;
    const float* q  = (const float*)d_in[0];
    const float* k  = (const float*)d_in[1];
    const float* v  = (const float*)d_in[2];
    const float* Wq = (const float*)d_in[3];
    const float* bq = (const float*)d_in[4];
    const float* Wk = (const float*)d_in[5];
    const float* bk = (const float*)d_in[6];
    const float* Wv = (const float*)d_in[7];
    const float* bv = (const float*)d_in[8];
    const float* Wo = (const float*)d_in[9];
    const float* bo = (const float*)d_in[10];
    float* out = (float*)d_out;

    const size_t elems = (size_t)M * DM;  // 8M elements
    __bf16* Qb  = (__bf16*)d_ws;
    __bf16* Kb  = Qb + elems;
    __bf16* Vb  = Kb + elems;
    __bf16* Ctx = Vb + elems;

    dim3 gemm_grid(DM / 128, M / 128);  // (8, 64)
    gemm_bias_kernel<float, __bf16, true>
        <<<gemm_grid, 256, 0, stream>>>(q, Wq, bq, Qb, M, DM, DM);
    gemm_bias_kernel<float, __bf16, true>
        <<<gemm_grid, 256, 0, stream>>>(k, Wk, bk, Kb, M, DM, DM);
    gemm_bias_kernel<float, __bf16, true>
        <<<gemm_grid, 256, 0, stream>>>(v, Wv, bv, Vb, M, DM, DM);

    dim3 attn_grid(S / 128, 16 /*heads*/, B);  // (16, 16, 4)
    flash_attn_kernel<<<attn_grid, 256, 0, stream>>>(Qb, Kb, Vb, Ctx);

    gemm_bias_kernel<__bf16, float, false>
        <<<gemm_grid, 256, 0, stream>>>(Ctx, Wo, bo, out, M, DM, DM);
}
